// BHS_GAT_16724602651177
// MI455X (gfx1250) — matrix operand order repaired and checked by reading, not compiled, not run on hardware
//
#include <hip/hip_runtime.h>
#include <hip/hip_bf16.h>

typedef float v8f   __attribute__((ext_vector_type(8)));
typedef __bf16 v8bf  __attribute__((ext_vector_type(8)));
typedef __bf16 v16bf __attribute__((ext_vector_type(16)));

#define BN    8192
#define HEADS 8
#define NSLOPE 0.2f

__device__ __forceinline__ float lrelu(float x) { return x > 0.f ? x : NSLOPE * x; }

// monotone float <-> uint mapping so unsigned atomicMax == float max
__device__ __forceinline__ unsigned f2mono(float f) {
    unsigned u = __float_as_uint(f);
    return (u & 0x80000000u) ? ~u : (u | 0x80000000u);
}
__device__ __forceinline__ float mono2f(unsigned u) {
    return __uint_as_float((u & 0x80000000u) ? (u & 0x7FFFFFFFu) : ~u);
}
__device__ __forceinline__ unsigned short f2bf(float f) {  // RNE fp32->bf16
    unsigned u = __float_as_uint(f);
    u += 0x7FFFu + ((u >> 16) & 1u);
    return (unsigned short)(u >> 16);
}

// ---------------- conv1: h = x@W1 (K=16), alpha_src/dst, init seg-max -------
__global__ void k_conv1_transform(const float* __restrict__ x, const float* __restrict__ W1,
                                  const float* __restrict__ a_src, const float* __restrict__ a_dst,
                                  float* __restrict__ ht, float* __restrict__ as_,
                                  float* __restrict__ ad_, unsigned* __restrict__ mmax) {
    __shared__ float xs[16], ps[64], pd[64];
    int n = blockIdx.x, t = threadIdx.x;
    if (t < 16) xs[t] = x[n * 16 + t];
    __syncthreads();
    float acc = 0.f;
#pragma unroll
    for (int k = 0; k < 16; ++k) acc += xs[k] * W1[k * 64 + t];
    ht[(size_t)n * 64 + t] = acc;
    int h = t >> 3, c = t & 7;
    ps[t] = acc * a_src[h * 8 + c];
    pd[t] = acc * a_dst[h * 8 + c];
    __syncthreads();
    if (c == 0) {
        float s = 0.f, d = 0.f;
#pragma unroll
        for (int i = 0; i < 8; ++i) { s += ps[h * 8 + i]; d += pd[h * 8 + i]; }
        as_[n * 8 + h] = s; ad_[n * 8 + h] = d;
        mmax[n * 8 + h] = f2mono(lrelu(s + d));   // self-loop seeds the segment max
    }
}

// ---------------- conv2 alpha: einsum over C=128 + init seg-max -------------
__global__ void k_conv2_alpha(const float* __restrict__ ht, const float* __restrict__ a_src,
                              const float* __restrict__ a_dst, float* __restrict__ as_,
                              float* __restrict__ ad_, unsigned* __restrict__ mmax) {
    __shared__ float ps[256], pd[256];
    int n = blockIdx.x, t = threadIdx.x;
    const float* row = ht + (size_t)n * 1024;
    int j0 = t * 4, h = j0 >> 7;
    float4 v4 = *(const float4*)(row + j0);
    const float* asr = a_src + h * 128, *adr = a_dst + h * 128;
    int c0 = j0 & 127;
    float s = v4.x * asr[c0] + v4.y * asr[c0 + 1] + v4.z * asr[c0 + 2] + v4.w * asr[c0 + 3];
    float d = v4.x * adr[c0] + v4.y * adr[c0 + 1] + v4.z * adr[c0 + 2] + v4.w * adr[c0 + 3];
    ps[t] = s; pd[t] = d;
    __syncthreads();
    if (t < 8) {
        float ss = 0.f, dd = 0.f;
#pragma unroll
        for (int i = 0; i < 32; ++i) { ss += ps[t * 32 + i]; dd += pd[t * 32 + i]; }
        as_[n * 8 + t] = ss; ad_[n * 8 + t] = dd;
        mmax[n * 8 + t] = f2mono(lrelu(ss + dd));
    }
}

// ---------------- edge passes: segment max / exp-sum ------------------------
__global__ void k_edge_max(const float* __restrict__ as_, const float* __restrict__ ad_,
                           const int* __restrict__ src, const int* __restrict__ dst,
                           unsigned* __restrict__ mmax, int E) {
    int t = blockIdx.x * blockDim.x + threadIdx.x;
    if (t >= E * 8) return;
    int e = t >> 3, h = t & 7;
    int s = src[e], d = dst[e];
    atomicMax(&mmax[d * 8 + h], f2mono(lrelu(as_[s * 8 + h] + ad_[d * 8 + h])));
}

__global__ void k_seg_init_sum(const float* __restrict__ as_, const float* __restrict__ ad_,
                               const unsigned* __restrict__ mmax, float* __restrict__ ssum, int NH) {
    int t = blockIdx.x * blockDim.x + threadIdx.x;
    if (t >= NH) return;
    ssum[t] = __expf(lrelu(as_[t] + ad_[t]) - mono2f(mmax[t]));   // self-loop term
}

__global__ void k_edge_expsum(const float* __restrict__ as_, const float* __restrict__ ad_,
                              const int* __restrict__ src, const int* __restrict__ dst,
                              const unsigned* __restrict__ mmax, float* __restrict__ ssum, int E) {
    int t = blockIdx.x * blockDim.x + threadIdx.x;
    if (t >= E * 8) return;
    int e = t >> 3, h = t & 7;
    int s = src[e], d = dst[e];
    float ev = lrelu(as_[s * 8 + h] + ad_[d * 8 + h]);
    atomicAdd(&ssum[d * 8 + h], __expf(ev - mono2f(mmax[d * 8 + h])));
}

// ---------------- aggregation: self-loop init + edge scatter-add ------------
__global__ void k_agg_init(const float* __restrict__ ht, const float* __restrict__ as_,
                           const float* __restrict__ ad_, const unsigned* __restrict__ mmax,
                           const float* __restrict__ ssum, float* __restrict__ out,
                           int HCshift, int Cshift, int total) {
    int t = blockIdx.x * blockDim.x + threadIdx.x;
    if (t >= total) return;
    int n = t >> HCshift, j = t & ((1 << HCshift) - 1), h = j >> Cshift;
    float e = lrelu(as_[n * 8 + h] + ad_[n * 8 + h]);
    float alpha = __expf(e - mono2f(mmax[n * 8 + h])) / (ssum[n * 8 + h] + 1e-16f);
    out[t] = ht[t] * alpha;
}

__global__ void k_edge_agg(const float* __restrict__ ht, const float* __restrict__ as_,
                           const float* __restrict__ ad_, const unsigned* __restrict__ mmax,
                           const float* __restrict__ ssum, const int* __restrict__ src,
                           const int* __restrict__ dst, float* __restrict__ out,
                           int HC, int Cshift) {
    __shared__ float wt[8];
    int e = blockIdx.x, t = threadIdx.x;
    int s = src[e], d = dst[e];
    if (t < 8) {
        float ev = lrelu(as_[s * 8 + t] + ad_[d * 8 + t]);
        wt[t] = __expf(ev - mono2f(mmax[d * 8 + t])) / (ssum[d * 8 + t] + 1e-16f);
    }
    __syncthreads();
    const float* hs = ht + (size_t)s * HC;
    float* od = out + (size_t)d * HC;
    for (int j = t; j < HC; j += blockDim.x)
        atomicAdd(&od[j], hs[j] * wt[j >> Cshift]);
}

// ---------------- bias + relu (+bf16 pack for WMMA A) -----------------------
__global__ void k_bias_relu_bf16(const float* __restrict__ in, const float* __restrict__ b,
                                 unsigned short* __restrict__ obf, int mask, int total) {
    int t = blockIdx.x * blockDim.x + threadIdx.x;
    if (t >= total) return;
    obf[t] = f2bf(fmaxf(in[t] + b[t & mask], 0.f));
}
__global__ void k_bias_relu_inplace(float* __restrict__ io, const float* __restrict__ b,
                                    int mask, int total) {
    int t = blockIdx.x * blockDim.x + threadIdx.x;
    if (t >= total) return;
    io[t] = fmaxf(io[t] + b[t & mask], 0.f);
}

// W2 [64,1024] fp32 -> W2^T [1024,64] bf16 (so each WMMA B-lane reads contiguous K)
__global__ void k_w2t_bf16(const float* __restrict__ W2, unsigned short* __restrict__ wt) {
    int t = blockIdx.x * blockDim.x + threadIdx.x;
    if (t >= 64 * 1024) return;
    int k = t >> 10, n = t & 1023;
    wt[n * 64 + k] = f2bf(W2[t]);
}

// ---------------- conv2 GEMM: [8192,64]bf16 @ [64,1024]bf16 -> fp32 ---------
// One wave computes a 16x64 output strip: 1 M-tile x 4 N-tiles. The A fragment
// (both K=32 halves) is loaded once and reused across the 4 N-tiles; K=64 is
// two v_wmma_f32_16x16x32_bf16 steps per tile -> 8 wmma per wave.
__global__ void k_gemm_bf16_wmma(const unsigned short* __restrict__ Abf,  // [8192,64]
                                 const unsigned short* __restrict__ BTbf, // [1024,64]
                                 float* __restrict__ D) {
    int wave = blockIdx.x * (blockDim.x >> 5) + (threadIdx.x >> 5);
    int lane = threadIdx.x & 31;
    int mt = wave >> 4, ng = wave & 15;            // 512 m-tiles x 16 n-groups(x4 tiles)
    if (mt >= 512) return;
    int lo16 = lane & 15, grp = lane >> 4;

    // A 16x32 bf16 fragment layout: elems 0..7 -> K=kb+grp*8+j ; 8..15 -> +16
    const unsigned short* arow = Abf + (size_t)(mt * 16 + lo16) * 64;
    v8bf a0 = *(const v8bf*)(arow + grp * 8);
    v8bf a1 = *(const v8bf*)(arow + grp * 8 + 16);
    v8bf a2 = *(const v8bf*)(arow + 32 + grp * 8);
    v8bf a3 = *(const v8bf*)(arow + 32 + grp * 8 + 16);
    v16bf A0{}, A1{};
#pragma unroll
    for (int i = 0; i < 8; ++i) { A0[i] = a0[i]; A0[i + 8] = a1[i]; A1[i] = a2[i]; A1[i + 8] = a3[i]; }

    v8f acc[4] = {v8f{}, v8f{}, v8f{}, v8f{}};
#pragma unroll
    for (int q = 0; q < 4; ++q) {
        int nt = ng * 4 + q;
        const unsigned short* brow = BTbf + (size_t)(nt * 16 + lo16) * 64; // col n = lane&15
        // B 32x16 fragment: elems 0..7 -> K = kb + grp*8 + j ; 8..15 -> +16 (matches A)
        v8bf b0 = *(const v8bf*)(brow + grp * 8);
        v8bf b1 = *(const v8bf*)(brow + grp * 8 + 16);
        v8bf b2 = *(const v8bf*)(brow + 32 + grp * 8);
        v8bf b3 = *(const v8bf*)(brow + 32 + grp * 8 + 16);
        v16bf B0{}, B1{};
#pragma unroll
        for (int i = 0; i < 8; ++i) { B0[i] = b0[i]; B0[i + 8] = b1[i]; B1[i] = b2[i]; B1[i + 8] = b3[i]; }
        acc[q] = __builtin_amdgcn_wmma_f32_16x16x32_bf16(false, A0, false, B0, (short)0, acc[q], false, false);
        acc[q] = __builtin_amdgcn_wmma_f32_16x16x32_bf16(false, A1, false, B1, (short)0, acc[q], false, false);
    }

    // D 16x16 f32: lane n = lane&15, VGPR r -> row m = grp*8 + r
#pragma unroll
    for (int q = 0; q < 4; ++q) {
        float* drow = D + (size_t)(mt * 16 + grp * 8) * 1024 + (ng * 4 + q) * 16 + lo16;
#pragma unroll
        for (int r = 0; r < 8; ++r) drow[(size_t)r * 1024] = acc[q][r];
    }
}

// ---------------- head GEMMs: [8, 1M] @ [1M, N] (N = 16 or 64) --------------
// Weight-streaming bound: coalesced w reads, K-chunk per block, LDS combine,
// global atomic-add into the tiny output.
template <int N>
__global__ void k_head_gemm(const float* __restrict__ h, const float* __restrict__ w,
                            float* __restrict__ out, int K, int KB) {
    const int RPI = 256 / N;                // k-rows advanced per loop iter
    __shared__ float red[256];
    int n = threadIdx.x % N, r = threadIdx.x / N;
    int k0 = blockIdx.x * KB;
    float acc[8] = {0.f, 0.f, 0.f, 0.f, 0.f, 0.f, 0.f, 0.f};
    for (int kk = k0 + r; kk < k0 + KB; kk += RPI) {
        if (kk + 8 * RPI < K) __builtin_prefetch(&w[(size_t)(kk + 8 * RPI) * N + n], 0, 1);
        float wv = w[(size_t)kk * N + n];   // contiguous across the block
#pragma unroll
        for (int b = 0; b < 8; ++b) acc[b] += h[(size_t)b * K + kk] * wv;
    }
#pragma unroll
    for (int b = 0; b < 8; ++b) {
        red[threadIdx.x] = acc[b];
        __syncthreads();
        if (r == 0) {
            float tot = 0.f;
            for (int i = 0; i < RPI; ++i) tot += red[i * N + n];
            atomicAdd(&out[b * N + n], tot);
        }
        __syncthreads();
    }
}

__global__ void k_zero(float* __restrict__ p, int count) {
    int t = blockIdx.x * blockDim.x + threadIdx.x;
    if (t < count) p[t] = 0.f;
}

// ---------------- final dueling head (1 block) ------------------------------
__global__ void k_head_final(const float* __restrict__ adv_pre, const float* __restrict__ adv_b,
                             const float* __restrict__ v1_pre, const float* __restrict__ v1b,
                             const float* __restrict__ w2, const float* __restrict__ b2,
                             const float* __restrict__ w3, const float* __restrict__ b3,
                             float* __restrict__ out) {
    __shared__ float v1[8 * 64], v2[8 * 64], vsc[8], advs[128];
    int t = threadIdx.x, b = t >> 6, n = t & 63;
    v1[b * 64 + n] = fmaxf(v1_pre[b * 64 + n] + v1b[n], 0.f);
    __syncthreads();
    float acc = b2[n];
    for (int k = 0; k < 64; ++k) acc += v1[b * 64 + k] * w2[k * 64 + n];
    v2[b * 64 + n] = fmaxf(acc, 0.f);
    __syncthreads();
    if (n == 0) {
        float a = b3[0];
        for (int k = 0; k < 64; ++k) a += v2[b * 64 + k] * w3[k];
        vsc[b] = a;
    }
    if (t < 128) advs[t] = fmaxf(adv_pre[t] + adv_b[t & 15], 0.f);
    __syncthreads();
    if (t < 128) {
        int base = t & ~3;
        float mean = 0.25f * (advs[base] + advs[base + 1] + advs[base + 2] + advs[base + 3]);
        out[t] = vsc[t >> 4] + advs[t] - mean;
    }
}

// ============================================================================
extern "C" void kernel_launch(void* const* d_in, const int* in_sizes, int n_in,
                              void* d_out, int out_size, void* d_ws, size_t ws_size,
                              hipStream_t stream) {
    const float* x      = (const float*)d_in[0];
    const int*   edge   = (const int*)d_in[1];
    const float* W1     = (const float*)d_in[2];
    const float* a_src1 = (const float*)d_in[3];
    const float* a_dst1 = (const float*)d_in[4];
    const float* b1     = (const float*)d_in[5];
    const float* W2     = (const float*)d_in[6];
    const float* a_src2 = (const float*)d_in[7];
    const float* a_dst2 = (const float*)d_in[8];
    const float* b2     = (const float*)d_in[9];
    const float* adv_w  = (const float*)d_in[10];
    const float* adv_b  = (const float*)d_in[11];
    const float* val1_w = (const float*)d_in[12];
    const float* val1_b = (const float*)d_in[13];
    const float* val2_w = (const float*)d_in[14];
    const float* val2_b = (const float*)d_in[15];
    const float* val3_w = (const float*)d_in[16];
    const float* val3_b = (const float*)d_in[17];
    float* out = (float*)d_out;

    const int E = in_sizes[1] / 2;           // 131072
    const int* src = edge;
    const int* dst = edge + E;

    // -------- workspace carve-out --------
    char* base = (char*)d_ws;
    auto alloc = [&](size_t bytes) { char* p = base; base += (bytes + 255) & ~(size_t)255; return p; };
    float*          ht1  = (float*)alloc((size_t)BN * 64 * 4);
    float*          as1  = (float*)alloc((size_t)BN * 8 * 4);
    float*          ad1  = (float*)alloc((size_t)BN * 8 * 4);
    unsigned*       m1   = (unsigned*)alloc((size_t)BN * 8 * 4);
    float*          s1   = (float*)alloc((size_t)BN * 8 * 4);
    float*          out1 = (float*)alloc((size_t)BN * 64 * 4);
    unsigned short* h1bf = (unsigned short*)alloc((size_t)BN * 64 * 2);
    unsigned short* w2t  = (unsigned short*)alloc((size_t)1024 * 64 * 2);
    float*          ht2  = (float*)alloc((size_t)BN * 1024 * 4);
    float*          as2  = (float*)alloc((size_t)BN * 8 * 4);
    float*          ad2  = (float*)alloc((size_t)BN * 8 * 4);
    unsigned*       m2   = (unsigned*)alloc((size_t)BN * 8 * 4);
    float*          s2   = (float*)alloc((size_t)BN * 8 * 4);
    float*          out2 = (float*)alloc((size_t)BN * 1024 * 4);
    float*          advp = (float*)alloc(128 * 4);
    float*          v1p  = (float*)alloc(512 * 4);
    (void)ws_size; (void)n_in; (void)out_size;

    const int EH_blocks = (E * 8 + 255) / 256;

    // ---------------- conv1 ----------------
    k_conv1_transform<<<BN, 64, 0, stream>>>(x, W1, a_src1, a_dst1, ht1, as1, ad1, m1);
    k_edge_max<<<EH_blocks, 256, 0, stream>>>(as1, ad1, src, dst, m1, E);
    k_seg_init_sum<<<(BN * 8 + 255) / 256, 256, 0, stream>>>(as1, ad1, m1, s1, BN * 8);
    k_edge_expsum<<<EH_blocks, 256, 0, stream>>>(as1, ad1, src, dst, m1, s1, E);
    k_agg_init<<<(BN * 64) / 256, 256, 0, stream>>>(ht1, as1, ad1, m1, s1, out1, 6, 3, BN * 64);
    k_edge_agg<<<E, 64, 0, stream>>>(ht1, as1, ad1, m1, s1, src, dst, out1, 64, 3);
    k_bias_relu_bf16<<<(BN * 64) / 256, 256, 0, stream>>>(out1, b1, h1bf, 63, BN * 64);

    // ---------------- conv2 ----------------
    k_w2t_bf16<<<(64 * 1024) / 256, 256, 0, stream>>>(W2, w2t);
    k_gemm_bf16_wmma<<<1024, 256, 0, stream>>>(h1bf, w2t, ht2);   // 8192 waves, 16x64 strip each
    k_conv2_alpha<<<BN, 256, 0, stream>>>(ht2, a_src2, a_dst2, as2, ad2, m2);
    k_edge_max<<<EH_blocks, 256, 0, stream>>>(as2, ad2, src, dst, m2, E);
    k_seg_init_sum<<<(BN * 8 + 255) / 256, 256, 0, stream>>>(as2, ad2, m2, s2, BN * 8);
    k_edge_expsum<<<EH_blocks, 256, 0, stream>>>(as2, ad2, src, dst, m2, s2, E);
    k_agg_init<<<(BN * 1024) / 256, 256, 0, stream>>>(ht2, as2, ad2, m2, s2, out2, 10, 7, BN * 1024);
    k_edge_agg<<<E, 256, 0, stream>>>(ht2, as2, ad2, m2, s2, src, dst, out2, 1024, 7);
    k_bias_relu_inplace<<<(BN * 1024) / 256, 256, 0, stream>>>(out2, b2, 1023, BN * 1024);
    // out2 is now h: [8, 1048576] row-major (node-major within batch)

    // ---------------- dueling head ----------------
    k_zero<<<1, 256, 0, stream>>>(advp, 128);
    k_zero<<<2, 256, 0, stream>>>(v1p, 512);
    const int K = 1024 * 1024, KB = 4096;                         // 256 K-chunks
    k_head_gemm<16><<<K / KB, 256, 0, stream>>>(out2, adv_w, advp, K, KB);
    k_head_gemm<64><<<K / KB, 256, 0, stream>>>(out2, val1_w, v1p, K, KB);
    k_head_final<<<1, 512, 0, stream>>>(advp, adv_b, v1p, val1_b, val2_w, val2_b,
                                        val3_w, val3_b, out);
}
